// LabelPropagation_50706383896899
// MI455X (gfx1250) — matrix operand order repaired and checked by reading, not compiled, not run on hardware
//
#include <hip/hip_runtime.h>

// Label propagation, restructured: sim·p = diag(invnorm) · F · (Fᵀ·p).
// B=8192, D=64, C=16.  All matmuls via v_wmma_f32_16x16x32_bf16, with every
// WMMA operand stored pre-swizzled in a WMMA lane layout (k = 16*half + e) so the hot
// loops issue only wide (32B/lane) vector loads.

#define B_N   8192
#define D_N   64
#define C_N   16
#define ALPHA 0.1f
#define ITERS 100
#define NPA   64   // blocks for the G = Fᵀ·P partial kernel
#define CPB   4    // K-chunks (32 rows) per block: 8192/32/NPA

typedef __bf16 v16bf __attribute__((ext_vector_type(16)));
typedef __bf16 v8bf  __attribute__((ext_vector_type(8)));
typedef float  v8f   __attribute__((ext_vector_type(8)));

// --- 16-bit WMMA operand layouts used here (wave32) -------------------------------
// A (16x32): lane = half*16 + m;  element e holds K = half*16 + e
//   inverse for k:  half = k>>4,      e = k&15
// B (32x16): lane = khalf*16 + n; element e holds K = khalf*16 + e
//   inverse for k:  khalf = k>>4,    e = k&15
// D (16x16 f32): lane = m8*16 + n (m8=m/8 selector); VGPR r holds M = r + 8*m8
// Tile storage: 512 halfs per tile, flat index = lane*16 + e  (32B per lane).

// ---------------------------------------------------------------------------
// Setup (once): invnorm, pre-swizzled A-operand copies of F, P0 = lab*m.
// ---------------------------------------------------------------------------
__global__ void lp_setup(const float* __restrict__ F,
                         const int*   __restrict__ labels,
                         const int*   __restrict__ mask,
                         float*  __restrict__ P,
                         __bf16* __restrict__ PhS,   // B-layout tiles of P
                         __bf16* __restrict__ FA,    // A-layout tiles of F   (update GEMM)
                         __bf16* __restrict__ FTA,   // A-layout tiles of Fᵀ  (reduction GEMM)
                         float*  __restrict__ invnorm) {
  int b = blockIdx.x * blockDim.x + threadIdx.x;
  if (b >= B_N) return;

  const int cb    = b >> 5;              // 32-row K-chunk of the reduction GEMM
  const int kk_a  = b & 31;              // k within that chunk
  const int haA   = kk_a >> 4;           // A-layout half for FTA
  const int eA    = kk_a & 15;
  const int tile  = b >> 4;              // 16-row M-tile of the update GEMM
  const int mA    = b & 15;

  float ss = 0.0f;
  #pragma unroll
  for (int d = 0; d < D_N; ++d) {
    float f = F[b * D_N + d];
    ss += f * f;
    __bf16 h = (__bf16)f;
    // FTA: A[m][k] = F[cb*32+k][w*16+m]   -> tile (cb*4 + w)
    {
      int w = d >> 4, m = d & 15;
      FTA[(size_t)(cb * 4 + w) * 512 + (haA * 16 + m) * 16 + eA] = h;
    }
    // FA: A[m][k] = F[tile*16+m][kc*32+k] -> tile (tile*2 + kc)
    {
      int kc = d >> 5, kk = d & 31;
      int ha = kk >> 4;
      int e  = kk & 15;
      FA[(size_t)(tile * 2 + kc) * 512 + (ha * 16 + mA) * 16 + e] = h;
    }
  }
  invnorm[b] = rsqrtf(ss);

  const int khB = (b >> 4) & 1;          // B-layout khalf for PhS
  const int eB  = b & 15;
  float mv = (float)mask[b];
  #pragma unroll
  for (int c = 0; c < C_N; ++c) {
    float p0 = (float)labels[b * C_N + c] * mv;
    P[b * C_N + c] = p0;
    PhS[(size_t)cb * 512 + (khB * 16 + c) * 16 + eB] = (__bf16)p0;
  }
}

// ---------------------------------------------------------------------------
// Kernel A: partial G = Fᵀ·P.  4 waves/block, one 16-wide d-tile each,
// CPB K-chunks of 32.  Per chunk: one 32B A-load, one 32B B-load, one WMMA.
// ---------------------------------------------------------------------------
__global__ void lp_gpart(const __bf16* __restrict__ FTA,
                         const __bf16* __restrict__ PhS,
                         float* __restrict__ Gpart) {
  const int wave = threadIdx.x >> 5;     // 0..3
  const int lane = threadIdx.x & 31;

  v8f acc = {};
  for (int j = 0; j < CPB; ++j) {
    const int cb = blockIdx.x * CPB + j;
    v16bf a  = *reinterpret_cast<const v16bf*>(
                   FTA + ((size_t)(cb * 4 + wave) * 512 + lane * 16));
    v16bf bt = *reinterpret_cast<const v16bf*>(
                   PhS + ((size_t)cb * 512 + lane * 16));
    acc = __builtin_amdgcn_wmma_f32_16x16x32_bf16(
        false, a, false, bt, (short)0, acc, false, false);
  }
  const int l16 = lane & 15, half = lane >> 4;
  float* gp = Gpart + (size_t)blockIdx.x * (D_N * C_N);
  #pragma unroll
  for (int r = 0; r < 8; ++r) {
    int m = wave * 16 + r + 8 * half;    // D layout
    gp[m * C_N + l16] = acc[r];
  }
}

// ---------------------------------------------------------------------------
// Kernel B: reduce partial G (into bf16 B-layout LDS tiles), then
// P <- m*((1-a)P + a*invnorm*(F·G)) + lab*(1-m); also refresh PhS.
// 64 blocks x 8 waves; one 16x16 output tile per wave (K=64 = 2 WMMAs).
// ---------------------------------------------------------------------------
__global__ void lp_update(const float*  __restrict__ Gpart,
                          const __bf16* __restrict__ FA,
                          const int*    __restrict__ labels,
                          const int*    __restrict__ mask,
                          const float*  __restrict__ invnorm,
                          float*  __restrict__ P,
                          __bf16* __restrict__ PhS) {
  __shared__ __bf16 Ghs[2 * 512];        // two 32x16 B-layout tiles of G
  // Phase 1: fixed-order reduction of NPA partials, written straight into
  // the WMMA-B lane layout (deterministic, no atomics).
  for (int i = threadIdx.x; i < D_N * C_N; i += blockDim.x) {
    float s = 0.0f;
    for (int p = 0; p < NPA; ++p) s += Gpart[(size_t)p * (D_N * C_N) + i];
    int d = i >> 4, c = i & 15;
    int kc = d >> 5, kk = d & 31;
    Ghs[kc * 512 + ((kk >> 4) * 16 + c) * 16 + (kk & 15)] = (__bf16)s;
  }
  __syncthreads();

  const int wave  = threadIdx.x >> 5;    // 0..7
  const int lane  = threadIdx.x & 31;
  const int l16   = lane & 15;
  const int half  = lane >> 4;
  const int bbase = blockIdx.x * 128 + wave * 16;
  const int tile  = bbase >> 4;

  v8f acc = {};
  #pragma unroll
  for (int kc = 0; kc < 2; ++kc) {
    v16bf a  = *reinterpret_cast<const v16bf*>(
                   FA + ((size_t)(tile * 2 + kc) * 512 + lane * 16));
    v16bf bt = *reinterpret_cast<const v16bf*>(&Ghs[kc * 512 + lane * 16]);
    acc = __builtin_amdgcn_wmma_f32_16x16x32_bf16(
        false, a, false, bt, (short)0, acc, false, false);
  }

  // Element-wise propagation step on the D tile; rows m = r + 8*half are the
  // 8 consecutive b-rows [bbase + 8*half, bbase + 8*half + 7].
  v8bf pk;
  #pragma unroll
  for (int r = 0; r < 8; ++r) {
    int b = bbase + 8 * half + r;
    float mv = (float)mask[b];
    float lv = (float)labels[b * C_N + l16];
    float pn = mv * ((1.0f - ALPHA) * P[b * C_N + l16]
                     + ALPHA * invnorm[b] * acc[r])
             + lv * (1.0f - mv);
    P[b * C_N + l16] = pn;
    pk[r] = (__bf16)pn;
  }
  // Refresh PhS (B-layout): these 8 rows map to contiguous e = 8*half + r,
  // khalf = (bbase>>4)&1, chunk cb = bbase>>5  ->  single 16B store.
  size_t idx = (size_t)(bbase >> 5) * 512
             + ((((bbase >> 4) & 1) * 16 + l16) * 16) + 8 * half;
  *reinterpret_cast<v8bf*>(PhS + idx) = pk;
}

// ---------------------------------------------------------------------------
// Final: argmax over classes (first-max tie-break, matching jnp.argmax).
// ---------------------------------------------------------------------------
__global__ void lp_argmax(const float* __restrict__ P, float* __restrict__ out) {
  int b = blockIdx.x * blockDim.x + threadIdx.x;
  if (b >= B_N) return;
  float best = P[b * C_N];
  int bi = 0;
  #pragma unroll
  for (int c = 1; c < C_N; ++c) {
    float v = P[b * C_N + c];
    if (v > best) { best = v; bi = c; }
  }
  out[b] = (float)bi;
}

// ---------------------------------------------------------------------------
extern "C" void kernel_launch(void* const* d_in, const int* in_sizes, int n_in,
                              void* d_out, int out_size, void* d_ws, size_t ws_size,
                              hipStream_t stream) {
  const float* F      = (const float*)d_in[0];
  const int*   labels = (const int*)d_in[1];
  const int*   mask   = (const int*)d_in[2];

  char* ws = (char*)d_ws;                       // all offsets 256B-aligned
  float*  P       = (float*) (ws + 0);          // 8192*16*4   = 512 KB
  __bf16* PhS     = (__bf16*)(ws + 524288);     // 256 tiles*512 bf16 = 256 KB
  __bf16* FA      = (__bf16*)(ws + 786432);     // 1024 tiles*512 bf16 = 1 MB
  __bf16* FTA     = (__bf16*)(ws + 1835008);    // 1024 tiles*512 bf16 = 1 MB
  float*  invnorm = (float*) (ws + 2883584);    // 8192*4      = 32 KB
  float*  Gpart   = (float*) (ws + 2916352);    // 64*1024*4   = 256 KB

  lp_setup<<<B_N / 256, 256, 0, stream>>>(F, labels, mask, P, PhS, FA, FTA, invnorm);
  for (int t = 0; t < ITERS; ++t) {
    lp_gpart <<<NPA,       128, 0, stream>>>(FTA, PhS, Gpart);
    lp_update<<<B_N / 128, 256, 0, stream>>>(Gpart, FA, labels, mask, invnorm, P, PhS);
  }
  lp_argmax<<<B_N / 256, 256, 0, stream>>>(P, (float*)d_out);
}
